// AdvancedGNN_55765855371639
// MI455X (gfx1250) — matrix operand order repaired and checked by reading, not compiled, not run on hardware
//
#include <hip/hip_runtime.h>
#include <hip/hip_bf16.h>

// ---------------------------------------------------------------------------
// AdvancedGNN on MI455X (gfx1250).
// All dense matmuls: v_wmma_f32_16x16x32_bf16, wave32, one 16x16 D tile per
// wave, A/B pre-packed as bf16 in fragment-native layout so the inner loop is
// just 4x global_load_b128 + 1x wmma (no LDS, no per-element converts/guards).
// GAT edge-attention GEMM algebraically folded:
//   (ea_full @ We * att_edge).sum(-1) == ea_full @ (We @ att_e^T)   [E+N,8]
// and the self-loop mean-edge-attr term == segment-mean of per-edge scalars.
// ---------------------------------------------------------------------------

#define NODE_IN 75
#define EDGE_IN 12
#define HID     256
#define HID2    512
#define HEADS   8
#define FH      32
#define N_GIN   5
#define N_GAT   2
#define NGRAPH  1024

typedef __attribute__((ext_vector_type(16))) __bf16 v16bf;
typedef __attribute__((ext_vector_type(8)))  __bf16 v8bf;
typedef __attribute__((ext_vector_type(8)))  float  v8f;

__device__ __forceinline__ __bf16 f2bf(float f) {
  unsigned u = __builtin_bit_cast(unsigned, f);
  unsigned r = (u + 0x7fffu + ((u >> 16) & 1u)) >> 16;   // RNE
  unsigned short h = (unsigned short)r;
  return __builtin_bit_cast(__bf16, h);
}

__device__ __forceinline__ void atomicMaxF(float* addr, float val) {
  if (val >= 0.0f) atomicMax((int*)addr, __float_as_int(val));
  else             atomicMin((unsigned int*)addr, (unsigned int)__float_as_int(val));
}

// ---------------------------------------------------------------------------
// GEMM: C[M,N] = epi(A[M,K] @ B[K,N]).
//  A: bf16 row-major, row stride lda (lda % 8 == 0, zero-padded to Kp).
//  BT: bf16 transposed weights [Np][Kp], Kp % 32 == 0, Np % 16 == 0, padded 0.
//  epi(v) = (v + bias)*g + be, optional relu; dual fp32/bf16 output.
//  M must be a multiple of 64. Block = 128 thr = 4 waves, tile 64x16.
// Fragment layouts (ISA 7.12.2, wave32):
//  A 16x32: lane l (half 0) row l holds K{k0..k0+7, k0+16..k0+23};
//           lane l+16 (half 1) holds K{+8 of each}  -> two 16B loads.
//  B 32x16: lane l col l holds same K as A (K{k0+half*8..+7, +16..+23}) -> two 16B loads.
// ---------------------------------------------------------------------------
__global__ __launch_bounds__(128)
void gemm_wmma(const __bf16* __restrict__ A, int lda,
               const __bf16* __restrict__ BT,
               const float* __restrict__ bias, const float* __restrict__ g,
               const float* __restrict__ be,
               float* __restrict__ Cf, __bf16* __restrict__ Cb, int ldb,
               int M, int N, int Kp, int relu)
{
  const int tid  = threadIdx.x;
  const int wid  = tid >> 5;
  const int lane = tid & 31;
  const int l    = lane & 15;
  const int half = lane >> 4;

  const int arow = blockIdx.x * 64 + wid * 16 + l;   // M % 64 == 0 -> always valid
  const int bcol = blockIdx.y * 16 + l;              // valid: BT padded to Np

  const __bf16* ap = A  + (size_t)arow * lda + half * 8;
  const __bf16* bp = BT + (size_t)bcol * Kp  + half * 8;

  v8f acc = {};
  for (int k0 = 0; k0 < Kp; k0 += 32) {
    v8bf a0 = *(const v8bf*)(ap + k0);
    v8bf a1 = *(const v8bf*)(ap + k0 + 16);
    v8bf b0 = *(const v8bf*)(bp + k0);
    v8bf b1 = *(const v8bf*)(bp + k0 + 16);
    v16bf af = __builtin_shufflevector(a0, a1, 0,1,2,3,4,5,6,7,8,9,10,11,12,13,14,15);
    v16bf bf = __builtin_shufflevector(b0, b1, 0,1,2,3,4,5,6,7,8,9,10,11,12,13,14,15);
    acc = __builtin_amdgcn_wmma_f32_16x16x32_bf16(false, af, false, bf,
                                                  (short)0, acc, false, false);
  }

  const int col = blockIdx.y * 16 + l;
  if (col < N) {
    float gg = g    ? g[col]    : 1.0f;
    float bb = be   ? be[col]   : 0.0f;
    float bi = bias ? bias[col] : 0.0f;
#pragma unroll
    for (int r = 0; r < 8; ++r) {
      int row = blockIdx.x * 64 + wid * 16 + half * 8 + r;   // D row mapping
      float v = (acc[r] + bi) * gg + bb;
      if (relu) v = fmaxf(v, 0.0f);
      if (Cf) Cf[(size_t)row * N   + col] = v;
      if (Cb) Cb[(size_t)row * ldb + col] = f2bf(v);
    }
  }
}

// ------------------------------ packing ------------------------------------
__global__ void pack_a_bf16(const float* __restrict__ A, __bf16* __restrict__ Ab,
                            int M, int K, int Kp) {
  size_t i = (size_t)blockIdx.x * blockDim.x + threadIdx.x;
  size_t st = (size_t)gridDim.x * blockDim.x, total = (size_t)M * Kp;
  for (; i < total; i += st) {
    int row = (int)(i / Kp), k = (int)(i % Kp);
    float v = (k < K) ? A[(size_t)row * K + k] : 0.0f;
    Ab[i] = f2bf(v);
  }
}
__global__ void pack_w_bf16(const float* __restrict__ W, __bf16* __restrict__ WT,
                            int K, int N, int Kp, int Np) {
  size_t i = (size_t)blockIdx.x * blockDim.x + threadIdx.x;
  size_t st = (size_t)gridDim.x * blockDim.x, total = (size_t)Np * Kp;
  for (; i < total; i += st) {
    int n = (int)(i / Kp), k = (int)(i % Kp);
    float v = (n < N && k < K) ? W[(size_t)k * N + n] : 0.0f;
    WT[i] = f2bf(v);
  }
}

// ------------------------------- pointwise ---------------------------------
__global__ void fill_f(float* p, size_t n, float v) {
  size_t i = (size_t)blockIdx.x * blockDim.x + threadIdx.x;
  size_t st = (size_t)gridDim.x * blockDim.x;
  for (; i < n; i += st) p[i] = v;
}
__global__ void ew_add(float* a, const float* b, size_t n) {
  size_t i = (size_t)blockIdx.x * blockDim.x + threadIdx.x;
  size_t st = (size_t)gridDim.x * blockDim.x;
  for (; i < n; i += st) a[i] += b[i];
}
__global__ void ew_add_relu(float* h, const float* t, size_t n) {
  size_t i = (size_t)blockIdx.x * blockDim.x + threadIdx.x;
  size_t st = (size_t)gridDim.x * blockDim.x;
  for (; i < n; i += st) h[i] = fmaxf(h[i] + t[i], 0.0f);
}
__global__ void gin_combine_k(const float* __restrict__ h, const float* __restrict__ agg,
                              const float* __restrict__ eps, __bf16* __restrict__ zb, size_t n) {
  float c = 1.0f + *eps;
  size_t i = (size_t)blockIdx.x * blockDim.x + threadIdx.x;
  size_t st = (size_t)gridDim.x * blockDim.x;
  for (; i < n; i += st) zb[i] = f2bf(c * h[i] + agg[i]);
}

// ------------------------------- graph ops ---------------------------------
__global__ __launch_bounds__(256)
void bcast_add(float* __restrict__ h, __bf16* __restrict__ hb,
               const float* __restrict__ vn, const int* __restrict__ batch, int N) {
  int n = blockIdx.x, f = threadIdx.x;
  if (n >= N) return;
  size_t idx = (size_t)n * HID + f;
  float v = h[idx] + vn[(size_t)batch[n] * HID + f];
  h[idx] = v;
  if (hb) hb[idx] = f2bf(v);
}
__global__ __launch_bounds__(256)
void seg_sum_nodes(const float* __restrict__ h, const int* __restrict__ batch,
                   float* __restrict__ vt, int N) {
  int n = blockIdx.x, f = threadIdx.x;
  if (n < N) atomicAdd(&vt[(size_t)batch[n] * HID + f], h[(size_t)n * HID + f]);
}
__global__ __launch_bounds__(256)
void seg_sum_edges(const float* __restrict__ h, const int* __restrict__ src,
                   const int* __restrict__ dst, float* __restrict__ agg, int E) {
  int e = blockIdx.x, f = threadIdx.x;
  if (e < E) atomicAdd(&agg[(size_t)dst[e] * HID + f], h[(size_t)src[e] * HID + f]);
}

// ------------------------------- GAT pieces --------------------------------
__global__ void build_we_att(const float* __restrict__ We, const float* __restrict__ att_e,
                             float* __restrict__ we_att) {
  int i = blockIdx.x * blockDim.x + threadIdx.x;
  if (i >= HID * HEADS) return;
  int k = i >> 3, hd = i & 7;
  float s = 0.0f;
  for (int f = 0; f < FH; ++f) s += We[(size_t)k * HID + hd * FH + f] * att_e[hd * FH + f];
  we_att[(size_t)k * HEADS + hd] = s;
}
__global__ void att_reduce(const float* __restrict__ xh, const float* __restrict__ att_s,
                           const float* __restrict__ att_d, float* __restrict__ a_src,
                           float* __restrict__ a_dst, int N) {
  int i = blockIdx.x * blockDim.x + threadIdx.x;
  if (i >= N * HEADS) return;
  int n = i >> 3, hd = i & 7;
  const float* xr = xh + (size_t)n * HID + hd * FH;
  float s = 0.0f, d = 0.0f;
  for (int f = 0; f < FH; ++f) { float v = xr[f]; s += v * att_s[hd * FH + f]; d += v * att_d[hd * FH + f]; }
  a_src[i] = s; a_dst[i] = d;
}
__global__ void edge_cnt_aesum(const float* __restrict__ ae_edge, const int* __restrict__ dst,
                               float* __restrict__ ae_sum, float* __restrict__ cntf, int E) {
  int i = blockIdx.x * blockDim.x + threadIdx.x;
  if (i >= E * HEADS) return;
  int e = i >> 3, hd = i & 7;
  int d = dst[e];
  atomicAdd(&ae_sum[(size_t)d * HEADS + hd], ae_edge[i]);
  if (hd == 0) atomicAdd(&cntf[d], 1.0f);
}
__global__ void alpha_lrelu_max(const float* __restrict__ a_src, const float* __restrict__ a_dst,
                                const float* __restrict__ ae_edge, const float* __restrict__ ae_sum,
                                const float* __restrict__ cntf, const int* __restrict__ src,
                                const int* __restrict__ dst, float* __restrict__ exbuf,
                                float* __restrict__ amax, int E, int N) {
  int i = blockIdx.x * blockDim.x + threadIdx.x;
  if (i >= (E + N) * HEADS) return;
  int r = i >> 3, hd = i & 7;
  int sr, dr; float ae;
  if (r < E) { sr = src[r]; dr = dst[r]; ae = ae_edge[i]; }
  else { int n = r - E; sr = dr = n; ae = ae_sum[(size_t)n * HEADS + hd] / fmaxf(cntf[n], 1.0f); }
  float al = a_src[(size_t)sr * HEADS + hd] + a_dst[(size_t)dr * HEADS + hd] + ae;
  al = al > 0.0f ? al : 0.2f * al;
  exbuf[i] = al;
  atomicMaxF(&amax[(size_t)dr * HEADS + hd], al);
}
__global__ void alpha_exp_den(const int* __restrict__ dst, float* __restrict__ exbuf,
                              const float* __restrict__ amax, float* __restrict__ den,
                              int E, int N) {
  int i = blockIdx.x * blockDim.x + threadIdx.x;
  if (i >= (E + N) * HEADS) return;
  int r = i >> 3, hd = i & 7;
  int dr = (r < E) ? dst[r] : (r - E);
  float ex = expf(exbuf[i] - amax[(size_t)dr * HEADS + hd]);
  exbuf[i] = ex;
  atomicAdd(&den[(size_t)dr * HEADS + hd], ex);
}
__global__ __launch_bounds__(256)
void gat_scatter(const float* __restrict__ xh, const float* __restrict__ exbuf,
                 const float* __restrict__ den, const int* __restrict__ src,
                 const int* __restrict__ dst, float* __restrict__ out, int E, int N) {
  int r = blockIdx.x, f = threadIdx.x;
  if (r >= E + N) return;
  int sr, dr;
  if (r < E) { sr = src[r]; dr = dst[r]; } else { sr = dr = r - E; }
  int hd = f >> 5;
  float w = exbuf[(size_t)r * HEADS + hd] / den[(size_t)dr * HEADS + hd];
  atomicAdd(&out[(size_t)dr * HID + f], xh[(size_t)sr * HID + f] * w);
}
__global__ __launch_bounds__(256)
void gat_finalize(float* __restrict__ h, const float* __restrict__ out,
                  const float* __restrict__ bias, const float* __restrict__ g,
                  const float* __restrict__ be, int N) {
  __shared__ float red[256];
  int n = blockIdx.x, f = threadIdx.x;
  if (n >= N) return;
  float v = out[(size_t)n * HID + f] + bias[f];
  red[f] = v; __syncthreads();
  for (int s = 128; s > 0; s >>= 1) { if (f < s) red[f] += red[f + s]; __syncthreads(); }
  float mu = red[0] * (1.0f / HID);
  __syncthreads();
  float d = v - mu; red[f] = d * d; __syncthreads();
  for (int s = 128; s > 0; s >>= 1) { if (f < s) red[f] += red[f + s]; __syncthreads(); }
  float var = red[0] * (1.0f / HID);
  float ln = d * rsqrtf(var + 1e-5f) * g[f] + be[f];
  size_t idx = (size_t)n * HID + f;
  float t = ln + h[idx];
  h[idx] = t > 0.0f ? t : (expf(t) - 1.0f);
}

// -------------------------------- pooling ----------------------------------
__global__ __launch_bounds__(256)
void pool_scatter(const float* __restrict__ h, const int* __restrict__ batch,
                  float* __restrict__ sum_g, float* __restrict__ max_g,
                  float* __restrict__ cnt_g, int N) {
  int n = blockIdx.x, f = threadIdx.x;
  if (n >= N) return;
  int b = batch[n];
  float v = h[(size_t)n * HID + f];
  atomicAdd(&sum_g[(size_t)b * HID + f], v);
  atomicMaxF(&max_g[(size_t)b * HID + f], v);
  if (f == 0) atomicAdd(&cnt_g[b], 1.0f);
}
__global__ void pool_build(const float* __restrict__ sum_g, const float* __restrict__ max_g,
                           const float* __restrict__ cnt_g, float* __restrict__ pool, int G) {
  int i = blockIdx.x * blockDim.x + threadIdx.x;
  if (i >= G * 3 * HID) return;
  int g = i / (3 * HID), f = i % (3 * HID);
  float v;
  if (f < HID)          v = sum_g[(size_t)g * HID + f] / fmaxf(cnt_g[g], 1.0f);
  else if (f < 2 * HID) v = max_g[(size_t)g * HID + (f - HID)];
  else                  v = sum_g[(size_t)g * HID + (f - 2 * HID)];
  pool[i] = v;
}
__global__ void final_lin(const float* __restrict__ z, const float* __restrict__ W,
                          const float* __restrict__ b, float* __restrict__ out, int G) {
  int g = blockIdx.x * blockDim.x + threadIdx.x;
  if (g >= G) return;
  float s = b[0];
  for (int k = 0; k < HID / 2; ++k) s += z[(size_t)g * (HID / 2) + k] * W[k];
  out[g] = s;
}

// =============================== host side =================================
static inline void gemm(const __bf16* A, int lda, const __bf16* WT,
                        const float* bias, const float* g, const float* be,
                        float* Cf, __bf16* Cb, int ldb,
                        int M, int N, int Kp, int relu, hipStream_t s) {
  dim3 grid(M / 64, (N + 15) / 16);
  gemm_wmma<<<grid, 128, 0, s>>>(A, lda, WT, bias, g, be, Cf, Cb, ldb, M, N, Kp, relu);
}
static inline int gs(size_t n) { size_t b = (n + 255) / 256; return (int)(b > 16384 ? 16384 : b); }

extern "C" void kernel_launch(void* const* d_in, const int* in_sizes, int n_in,
                              void* d_out, int out_size, void* d_ws, size_t ws_size,
                              hipStream_t stream) {
  const int N = in_sizes[0] / NODE_IN;     // 65536
  const int E = in_sizes[1] / EDGE_IN;     // 262144
  const int G = NGRAPH;

  const float* x   = (const float*)d_in[0];
  const float* eat = (const float*)d_in[1];

  // ---- params: JAX pytree flatten order (dict keys sorted alphabetically) --
  int cur = 2;
  auto nx = [&]() { return (const float*)d_in[cur++]; };
  const float *ee_W = nx(), *ee_b = nx(), *ee_be = nx(), *ee_g = nx();
  struct GatP { const float *W,*We,*att_dst,*att_edge,*att_src,*bias,*ln_be,*ln_g; } gat[N_GAT];
  for (int i = 0; i < N_GAT; ++i) {
    gat[i].W = nx(); gat[i].We = nx(); gat[i].att_dst = nx(); gat[i].att_edge = nx();
    gat[i].att_src = nx(); gat[i].bias = nx(); gat[i].ln_be = nx(); gat[i].ln_g = nx();
  }
  struct GinP { const float *bn1_be,*bn1_g,*eps,*l1W,*l1b,*l2W,*l2b,*nm_be,*nm_g; } gin[N_GIN];
  for (int i = 0; i < N_GIN; ++i) {
    gin[i].bn1_be = nx(); gin[i].bn1_g = nx(); gin[i].eps = nx();
    gin[i].l1W = nx(); gin[i].l1b = nx(); gin[i].l2W = nx(); gin[i].l2b = nx();
    gin[i].nm_be = nx(); gin[i].nm_g = nx();
  }
  const float *ne_W = nx(), *ne_b = nx(), *ne_be = nx(), *ne_g = nx();
  const float *p_bn1be = nx(), *p_bn1g = nx(), *p_bn2be = nx(), *p_bn2g = nx();
  const float *p_bn3be = nx(), *p_bn3g = nx();
  const float *p_l1W = nx(), *p_l1b = nx(), *p_l2W = nx(), *p_l2b = nx();
  const float *p_l3W = nx(), *p_l3b = nx(), *p_l4W = nx(), *p_l4b = nx();
  struct VnP { const float *bn_be,*bn_g,*l1W,*l1b,*l2W,*l2b; } vn[N_GIN + N_GAT];
  for (int i = 0; i < N_GIN + N_GAT; ++i) {
    vn[i].bn_be = nx(); vn[i].bn_g = nx(); vn[i].l1W = nx(); vn[i].l1b = nx();
    vn[i].l2W = nx(); vn[i].l2b = nx();
  }
  const int* edge_index = (const int*)d_in[cur++];
  const int* batch      = (const int*)d_in[cur++];
  (void)n_in; (void)ws_size; (void)out_size;
  const int* src = edge_index;
  const int* dst = edge_index + E;

  // ---- workspace carve ----------------------------------------------------
  char* wp = (char*)d_ws;
  auto alloc = [&](size_t bytes) { void* p = wp; wp += (bytes + 255) & ~(size_t)255; return p; };
  // fp32
  float* h     = (float*)alloc((size_t)N * HID * 4);
  float* bufA  = (float*)alloc((size_t)N * HID * 4);   // xh
  float* bufC  = (float*)alloc((size_t)N * HID * 4);   // agg / l2-out / gat-out
  float* ae_edge = (float*)alloc((size_t)E * HEADS * 4);
  float* exbuf   = (float*)alloc((size_t)(E + N) * HEADS * 4);
  float* a_src = (float*)alloc((size_t)N * HEADS * 4);
  float* a_dst = (float*)alloc((size_t)N * HEADS * 4);
  float* ae_sum = (float*)alloc((size_t)N * HEADS * 4);
  float* amax   = (float*)alloc((size_t)N * HEADS * 4);
  float* den    = (float*)alloc((size_t)N * HEADS * 4);
  float* cntf   = (float*)alloc((size_t)N * 4);
  float* we_att = (float*)alloc((size_t)HID * HEADS * 4);
  float* virt = (float*)alloc((size_t)G * HID * 4);
  float* vt   = (float*)alloc((size_t)G * HID * 4);
  float* vo   = (float*)alloc((size_t)G * HID * 4);
  float* cnt_g = (float*)alloc((size_t)G * 4);
  float* sum_g = (float*)alloc((size_t)G * HID * 4);
  float* max_g = (float*)alloc((size_t)G * HID * 4);
  float* pool  = (float*)alloc((size_t)G * 3 * HID * 4);
  float* z3    = (float*)alloc((size_t)G * (HID / 2) * 4);
  // bf16 activations
  __bf16* abf   = (__bf16*)alloc((size_t)N * HID  * 2);  // x-packed / z / h (GAT)
  __bf16* hidbf = (__bf16*)alloc((size_t)N * HID2 * 2);  // GIN hidden; also edge_attr-packed
  __bf16* eabf  = (__bf16*)alloc((size_t)E * HID  * 2);  // encoded edge features
  __bf16* vtbf  = (__bf16*)alloc((size_t)G * HID  * 2);
  __bf16* vhbf  = (__bf16*)alloc((size_t)G * HID2 * 2);
  __bf16* poolbf = (__bf16*)alloc((size_t)G * 3 * HID * 2);
  __bf16* z1bf  = (__bf16*)alloc((size_t)G * HID2 * 2);
  __bf16* z2bf  = (__bf16*)alloc((size_t)G * HID  * 2);

  // ---- weight packs (bf16, transposed, padded) ----------------------------
  auto packW = [&](const float* W, int K, int Nn) -> const __bf16* {
    int Kp = (K + 31) & ~31, Np = (Nn + 15) & ~15;
    __bf16* dstp = (__bf16*)alloc((size_t)Np * Kp * 2);
    pack_w_bf16<<<gs((size_t)Np * Kp), 256, 0, stream>>>(W, dstp, K, Nn, Kp, Np);
    return dstp;
  };
  const __bf16* WTne = packW(ne_W, NODE_IN, HID);
  const __bf16* WTee = packW(ee_W, EDGE_IN, HID);
  const __bf16 *WTg1[N_GIN], *WTg2[N_GIN];
  for (int i = 0; i < N_GIN; ++i) { WTg1[i] = packW(gin[i].l1W, HID, HID2); WTg2[i] = packW(gin[i].l2W, HID2, HID); }
  const __bf16 *WTgw[N_GAT], *WTwa[N_GAT];
  for (int j = 0; j < N_GAT; ++j) {
    WTgw[j] = packW(gat[j].W, HID, HID);
    build_we_att<<<(HID * HEADS + 255) / 256, 256, 0, stream>>>(gat[j].We, gat[j].att_edge, we_att);
    WTwa[j] = packW(we_att, HID, HEADS);
  }
  const __bf16 *WTv1[N_GIN + N_GAT], *WTv2[N_GIN + N_GAT];
  for (int i = 0; i < N_GIN + N_GAT; ++i) { WTv1[i] = packW(vn[i].l1W, HID, HID2); WTv2[i] = packW(vn[i].l2W, HID2, HID); }
  const __bf16* WTp1 = packW(p_l1W, 3 * HID, HID2);
  const __bf16* WTp2 = packW(p_l2W, HID2, HID);
  const __bf16* WTp3 = packW(p_l3W, HID, HID / 2);

  const float NEG_INF = -3.3e38f;

  // ---- encoders -----------------------------------------------------------
  pack_a_bf16<<<gs((size_t)N * 96), 256, 0, stream>>>(x, abf, N, NODE_IN, 96);
  gemm(abf, 96, WTne, ne_b, ne_g, ne_be, h, nullptr, 0, N, HID, 96, 1, stream);
  pack_a_bf16<<<gs((size_t)E * 32), 256, 0, stream>>>(eat, hidbf, E, EDGE_IN, 32);
  gemm(hidbf, 32, WTee, ee_b, ee_g, ee_be, nullptr, eabf, HID, E, HID, 32, 1, stream);
  hipMemsetAsync(virt, 0, (size_t)G * HID * 4, stream);

  // ---- GIN stack ----------------------------------------------------------
  for (int i = 0; i < N_GIN; ++i) {
    bcast_add<<<N, 256, 0, stream>>>(h, nullptr, virt, batch, N);
    hipMemsetAsync(bufC, 0, (size_t)N * HID * 4, stream);
    seg_sum_edges<<<E, 256, 0, stream>>>(h, src, dst, bufC, E);
    gin_combine_k<<<gs((size_t)N * HID), 256, 0, stream>>>(h, bufC, gin[i].eps, abf, (size_t)N * HID);
    gemm(abf, HID, WTg1[i], gin[i].l1b, gin[i].bn1_g, gin[i].bn1_be,
         nullptr, hidbf, HID2, N, HID2, HID, 1, stream);
    gemm(hidbf, HID2, WTg2[i], gin[i].l2b, gin[i].nm_g, gin[i].nm_be,
         bufC, nullptr, 0, N, HID, HID2, 0, stream);
    ew_add_relu<<<gs((size_t)N * HID), 256, 0, stream>>>(h, bufC, (size_t)N * HID);
    // virtual node
    hipMemsetAsync(vt, 0, (size_t)G * HID * 4, stream);
    seg_sum_nodes<<<N, 256, 0, stream>>>(h, batch, vt, N);
    pack_a_bf16<<<gs((size_t)G * HID), 256, 0, stream>>>(vt, vtbf, G, HID, HID);
    gemm(vtbf, HID, WTv1[i], vn[i].l1b, vn[i].bn_g, vn[i].bn_be,
         nullptr, vhbf, HID2, G, HID2, HID, 1, stream);
    gemm(vhbf, HID2, WTv2[i], vn[i].l2b, nullptr, nullptr, vo, nullptr, 0, G, HID, HID2, 0, stream);
    ew_add<<<gs((size_t)G * HID), 256, 0, stream>>>(virt, vo, (size_t)G * HID);
  }

  // ---- GAT stack ----------------------------------------------------------
  for (int j = 0; j < N_GAT; ++j) {
    const GatP& P = gat[j];
    bcast_add<<<N, 256, 0, stream>>>(h, abf, virt, batch, N);
    gemm(eabf, HID, WTwa[j], nullptr, nullptr, nullptr, ae_edge, nullptr, 0, E, HEADS, HID, 0, stream);
    gemm(abf, HID, WTgw[j], nullptr, nullptr, nullptr, bufA, nullptr, 0, N, HID, HID, 0, stream);
    att_reduce<<<(N * HEADS + 255) / 256, 256, 0, stream>>>(bufA, P.att_src, P.att_dst, a_src, a_dst, N);
    hipMemsetAsync(ae_sum, 0, (size_t)N * HEADS * 4, stream);
    hipMemsetAsync(cntf, 0, (size_t)N * 4, stream);
    edge_cnt_aesum<<<(E * HEADS + 255) / 256, 256, 0, stream>>>(ae_edge, dst, ae_sum, cntf, E);
    fill_f<<<gs((size_t)N * HEADS), 256, 0, stream>>>(amax, (size_t)N * HEADS, NEG_INF);
    hipMemsetAsync(den, 0, (size_t)N * HEADS * 4, stream);
    int R = (E + N) * HEADS;
    alpha_lrelu_max<<<(R + 255) / 256, 256, 0, stream>>>(a_src, a_dst, ae_edge, ae_sum, cntf,
                                                         src, dst, exbuf, amax, E, N);
    alpha_exp_den<<<(R + 255) / 256, 256, 0, stream>>>(dst, exbuf, amax, den, E, N);
    hipMemsetAsync(bufC, 0, (size_t)N * HID * 4, stream);
    gat_scatter<<<E + N, 256, 0, stream>>>(bufA, exbuf, den, src, dst, bufC, E, N);
    gat_finalize<<<N, 256, 0, stream>>>(h, bufC, P.bias, P.ln_g, P.ln_be, N);
    // virtual node
    const VnP& V = vn[N_GIN + j];
    hipMemsetAsync(vt, 0, (size_t)G * HID * 4, stream);
    seg_sum_nodes<<<N, 256, 0, stream>>>(h, batch, vt, N);
    pack_a_bf16<<<gs((size_t)G * HID), 256, 0, stream>>>(vt, vtbf, G, HID, HID);
    gemm(vtbf, HID, WTv1[N_GIN + j], V.l1b, V.bn_g, V.bn_be,
         nullptr, vhbf, HID2, G, HID2, HID, 1, stream);
    gemm(vhbf, HID2, WTv2[N_GIN + j], V.l2b, nullptr, nullptr, vo, nullptr, 0, G, HID, HID2, 0, stream);
    ew_add<<<gs((size_t)G * HID), 256, 0, stream>>>(virt, vo, (size_t)G * HID);
  }

  // ---- pooling + prediction MLP ------------------------------------------
  hipMemsetAsync(cnt_g, 0, (size_t)G * 4, stream);
  hipMemsetAsync(sum_g, 0, (size_t)G * HID * 4, stream);
  fill_f<<<gs((size_t)G * HID), 256, 0, stream>>>(max_g, (size_t)G * HID, NEG_INF);
  pool_scatter<<<N, 256, 0, stream>>>(h, batch, sum_g, max_g, cnt_g, N);
  pool_build<<<(G * 3 * HID + 255) / 256, 256, 0, stream>>>(sum_g, max_g, cnt_g, pool, G);
  pack_a_bf16<<<gs((size_t)G * 3 * HID), 256, 0, stream>>>(pool, poolbf, G, 3 * HID, 3 * HID);
  gemm(poolbf, 3 * HID, WTp1, p_l1b, p_bn1g, p_bn1be, nullptr, z1bf, HID2, G, HID2, 3 * HID, 1, stream);
  gemm(z1bf, HID2, WTp2, p_l2b, p_bn2g, p_bn2be, nullptr, z2bf, HID, G, HID, HID2, 1, stream);
  gemm(z2bf, HID, WTp3, p_l3b, p_bn3g, p_bn3be, z3, nullptr, 0, G, HID / 2, HID, 1, stream);
  final_lin<<<(G + 255) / 256, 256, 0, stream>>>(z3, p_l4W, p_l4b, (float*)d_out, G);
}
